// EndpointDistanceLossAverage_53824530153737
// MI455X (gfx1250) — matrix operand order repaired and checked by reading, not compiled, not run on hardware
//
#include <hip/hip_runtime.h>
#include <math.h>

// ---------------------------------------------------------------------------
// EndpointDistanceLoss for MI455X (gfx1250, wave32)
//   - 40 fused erode/erode/dilate stencil iterations, 32x32 tiles + halo-3 in LDS
//   - tile staging via GLOBAL_LOAD_ASYNC_TO_LDS_B32 (ASYNCcnt), fallback plain
//   - eroded-img writeback via GLOBAL_STORE_ASYNC_FROM_LDS_B32 (img is in LDS)
//   - weighted-coordinate reduction via v_wmma_f32_16x16x32_f16 (fixed B,
//     per-chunk accumulators; global-x recovered in f32 epilogue)
//   - deterministic two-stage reductions (no atomics)
// ---------------------------------------------------------------------------

#define BATCH   8
#define HDIM    512
#define WDIM    512
#define HWSZ    (HDIM * WDIM)      // 262144
#define NIMG    16                 // 8 pred + 8 true
#define NITER   40
#define TILE    32
#define TPI     16                 // tiles per image edge (512/32)
#define NTILES  (TPI * TPI)        // 256 tiles per image

typedef _Float16 f16_t;
typedef f16_t v16h __attribute__((ext_vector_type(16)));
typedef float v8f  __attribute__((ext_vector_type(8)));

#define POS_INF  (__builtin_inff())
#define NEG_INF  (-__builtin_inff())

// --------- CDNA5 async global<->LDS paths (guarded; plain-load fallback) ----
#if defined(__has_builtin)
#  if __has_builtin(__builtin_amdgcn_global_load_async_to_lds_b32) && \
      __has_builtin(__builtin_amdgcn_s_wait_asynccnt)
#    define HAVE_ASYNC_LD 1
#  endif
#  if __has_builtin(__builtin_amdgcn_global_store_async_from_lds_b32) && \
      __has_builtin(__builtin_amdgcn_s_wait_asynccnt)
#    define HAVE_ASYNC_ST 1
#  endif
#endif
#ifndef HAVE_ASYNC_LD
#  define HAVE_ASYNC_LD 0
#endif
#ifndef HAVE_ASYNC_ST
#  define HAVE_ASYNC_ST 0
#endif

// Builtin signature uses int* in address spaces 1 (global) and 3 (LDS).
typedef __attribute__((address_space(1))) int as1_int;
typedef __attribute__((address_space(3))) int as3_int;

__device__ __forceinline__ void ldgsts_f32(const float* g, float* l) {
#if HAVE_ASYNC_LD
    __builtin_amdgcn_global_load_async_to_lds_b32(
        (as1_int*)g, (as3_int*)l, /*offset=*/0, /*cpol=*/0);
#else
    *l = *g;
#endif
}

__device__ __forceinline__ void stgasync_f32(float* g, const float* l) {
#if HAVE_ASYNC_ST
    __builtin_amdgcn_global_store_async_from_lds_b32(
        (as1_int*)g, (as3_int*)l, /*offset=*/0, /*cpol=*/0);
#else
    *g = *l;
#endif
}

__device__ __forceinline__ void wait_async_lds() {
#if HAVE_ASYNC_LD
    __builtin_amdgcn_s_wait_asynccnt(0);
#endif
}

// ---------------------------------------------------------------------------
// Kernel 1: softmax-fg + y_true cast + fused dice partial sums.
//   img[0..7]   = sigmoid(x1 - x0)       (softmax over 2 channels, fg)
//   img[8..15]  = (float)y_true
//   dicePart[blk][3] = { sum(y*p), sum(y), sum(p) }  (deterministic tree)
// ---------------------------------------------------------------------------
__global__ void prep_kernel(const float* __restrict__ no,
                            const int*   __restrict__ yt,
                            float* __restrict__ img,
                            float* __restrict__ dicePart) {
    const int tid = threadIdx.x;
    const int gstride = gridDim.x * blockDim.x;   // 1024*256 = 262144
    float s_inter = 0.f, s_y = 0.f, s_p = 0.f;
    for (int i = blockIdx.x * blockDim.x + tid; i < BATCH * HWSZ; i += gstride) {
        const int b = i / HWSZ;
        const int r = i - b * HWSZ;
        const float x0 = no[(size_t)b * 2 * HWSZ + r];
        const float x1 = no[(size_t)b * 2 * HWSZ + HWSZ + r];
        const float p  = 1.0f / (1.0f + expf(x0 - x1));   // softmax fg channel
        const float t  = (float)yt[i];
        img[(size_t)b * HWSZ + r]           = p;
        img[(size_t)(BATCH + b) * HWSZ + r] = t;
        s_inter += t * p;
        s_y     += t;
        s_p     += p;
    }
    __shared__ float sm[256];
    float vals[3] = { s_inter, s_y, s_p };
    for (int c = 0; c < 3; ++c) {
        sm[tid] = vals[c];
        __syncthreads();
        for (int s = 128; s > 0; s >>= 1) {
            if (tid < s) sm[tid] += sm[tid + s];
            __syncthreads();
        }
        if (tid == 0) dicePart[blockIdx.x * 3 + c] = sm[0];
        __syncthreads();
    }
}

// ---------------------------------------------------------------------------
// Kernel 2: skel = relu(img - dilate(erode(img)))   (halo 2)
//   erode = 5-point min cross (pad +inf), dilate = 3x3 max (pad -inf)
// ---------------------------------------------------------------------------
__global__ void init_skel_kernel(const float* __restrict__ img,
                                 float* __restrict__ skel) {
    const int tileId = blockIdx.x % NTILES;
    const int iidx   = blockIdx.x / NTILES;
    const int ty = (tileId / TPI) * TILE;
    const int tx = (tileId % TPI) * TILE;
    const size_t off = (size_t)iidx * HWSZ;
    const int tid = threadIdx.x;

    __shared__ float sIn[36 * 37];   // origin (-2,-2), stride 37
    __shared__ float sE1[34 * 37];   // origin (-1,-1)

    for (int i = tid; i < 36 * 36; i += 256) {
        const int ly = i / 36, lx = i - ly * 36;
        const int gy = ty + ly - 2, gx = tx + lx - 2;
        const bool inb = (gy >= 0) & (gy < HDIM) & (gx >= 0) & (gx < WDIM);
        if (inb) ldgsts_f32(&img[off + (size_t)gy * WDIM + gx], &sIn[ly * 37 + lx]);
        else     sIn[ly * 37 + lx] = POS_INF;
    }
    wait_async_lds();
    __syncthreads();

    for (int i = tid; i < 34 * 34; i += 256) {
        const int a = i / 34, b = i - a * 34;
        const int gy = ty + a - 1, gx = tx + b - 1;
        const bool inb = (gy >= 0) & (gy < HDIM) & (gx >= 0) & (gx < WDIM);
        const int c = (a + 1) * 37 + (b + 1);
        float m = sIn[c];
        m = fminf(m, sIn[c - 37]); m = fminf(m, sIn[c + 37]);
        m = fminf(m, sIn[c - 1]);  m = fminf(m, sIn[c + 1]);
        sE1[a * 37 + b] = inb ? m : NEG_INF;   // dilate pads with -inf
    }
    __syncthreads();

    const int lx = tid & 31;
    for (int k = 0; k < 4; ++k) {
        const int ry = (tid >> 5) + k * 8;
        float mx = NEG_INF;
        for (int dy = 0; dy < 3; ++dy)
            for (int dx = 0; dx < 3; ++dx)
                mx = fmaxf(mx, sE1[(ry + dy) * 37 + (lx + dx)]);
        const float c = sIn[(ry + 2) * 37 + (lx + 2)];
        skel[off + (size_t)(ty + ry) * WDIM + (tx + lx)] = fmaxf(c - mx, 0.0f);
    }
}

// ---------------------------------------------------------------------------
// Kernel 3: one soft_skel iteration, fully fused (halo 3):
//   e1 = erode(img)            -> new img (kept in LDS, async-stored out)
//   e2 = erode(e1); o = dilate(e2)
//   delta = relu(e1 - o); skel += relu(delta - skel*delta)
// ---------------------------------------------------------------------------
__global__ void iter_kernel(const float* __restrict__ src,
                            float* __restrict__ dst,
                            float* __restrict__ skel) {
    const int tileId = blockIdx.x % NTILES;
    const int iidx   = blockIdx.x / NTILES;
    const int ty = (tileId / TPI) * TILE;
    const int tx = (tileId % TPI) * TILE;
    const size_t off = (size_t)iidx * HWSZ;
    const int tid = threadIdx.x;

    __shared__ float sIn[38 * 39];   // origin (-3,-3), stride 39
    __shared__ float sE1[36 * 39];   // origin (-2,-2)
    __shared__ float sE2[34 * 39];   // origin (-1,-1)

    __builtin_prefetch(&skel[off + (size_t)ty * WDIM + tx], 0, 0); // global_prefetch_b8

    for (int i = tid; i < 38 * 38; i += 256) {
        const int ly = i / 38, lx = i - ly * 38;
        const int gy = ty + ly - 3, gx = tx + lx - 3;
        const bool inb = (gy >= 0) & (gy < HDIM) & (gx >= 0) & (gx < WDIM);
        if (inb) ldgsts_f32(&src[off + (size_t)gy * WDIM + gx], &sIn[ly * 39 + lx]);
        else     sIn[ly * 39 + lx] = POS_INF;
    }
    wait_async_lds();
    __syncthreads();

    for (int i = tid; i < 36 * 36; i += 256) {
        const int a = i / 36, b = i - a * 36;
        const int gy = ty + a - 2, gx = tx + b - 2;
        const bool inb = (gy >= 0) & (gy < HDIM) & (gx >= 0) & (gx < WDIM);
        const int c = (a + 1) * 39 + (b + 1);
        float m = sIn[c];
        m = fminf(m, sIn[c - 39]); m = fminf(m, sIn[c + 39]);
        m = fminf(m, sIn[c - 1]);  m = fminf(m, sIn[c + 1]);
        sE1[a * 39 + b] = inb ? m : POS_INF;   // feeds next erode: pad +inf
    }
    __syncthreads();

    for (int i = tid; i < 34 * 34; i += 256) {
        const int a = i / 34, b = i - a * 34;
        const int gy = ty + a - 1, gx = tx + b - 1;
        const bool inb = (gy >= 0) & (gy < HDIM) & (gx >= 0) & (gx < WDIM);
        const int c = (a + 1) * 39 + (b + 1);
        float m = sE1[c];
        m = fminf(m, sE1[c - 39]); m = fminf(m, sE1[c + 39]);
        m = fminf(m, sE1[c - 1]);  m = fminf(m, sE1[c + 1]);
        sE2[a * 39 + b] = inb ? m : NEG_INF;   // feeds dilate: pad -inf
    }
    __syncthreads();

    const int lx = tid & 31;
    for (int k = 0; k < 4; ++k) {
        const int ry = (tid >> 5) + k * 8;
        float mx = NEG_INF;
        for (int dy = 0; dy < 3; ++dy)
            for (int dx = 0; dx < 3; ++dx)
                mx = fmaxf(mx, sE2[(ry + dy) * 39 + (lx + dx)]);
        const int e1i = (ry + 2) * 39 + (lx + 2);
        const float imgNew = sE1[e1i];
        const float delta  = fmaxf(imgNew - mx, 0.0f);
        const size_t gi = off + (size_t)(ty + ry) * WDIM + (tx + lx);
        const float s = skel[gi];
        skel[gi] = s + fmaxf(delta - s * delta, 0.0f);
        // new img already lives in LDS (sE1): LDS -> global async store
        stgasync_f32(&dst[gi], &sE1[e1i]);
    }
    // async stores drained by S_ENDPGM's implicit wait-idle
}

// ---------------------------------------------------------------------------
// Kernel 4: soft_endpoints + weighted-coordinate reduction per 16-row strip.
//   nsum = 3x3 box sum + 9*center (zero pad); ep = exp(-(nsum-11)^2) * skel
//   Per strip: S = sum(ep), Sx = sum(ep*x), Sy = sum(ep*y)
//   D(16x16) += A(16x32,f16) * B(32x16,f16) via v_wmma_f32_16x16x32_f16.
//   B is chunk-invariant (col0 = 1, col1 = local k); per-chunk accumulators;
//   the global-x offset is applied exactly in the f32 epilogue.
// ---------------------------------------------------------------------------
__global__ void endpoint_kernel(const float* __restrict__ skel,
                                float* __restrict__ epPart) {
    const int strip = blockIdx.x % 32;       // 32 strips of 16 rows
    const int iidx  = blockIdx.x / 32;
    const int y0 = strip * 16;
    const size_t off = (size_t)iidx * HWSZ;
    const int tid = threadIdx.x;

    __shared__ float sS[18 * 514];            // rows y0-1..y0+16, cols -1..512
    __shared__ f16_t sEp[16 * 512];
    __shared__ float sRed[8][2][2][16];       // [wave][chunk][col 0|1][row]

    for (int i = tid; i < 18 * 514; i += 256) {
        const int ly = i / 514, lx = i - ly * 514;
        const int gy = y0 + ly - 1, gx = lx - 1;
        const bool inb = (gy >= 0) & (gy < HDIM) & (gx >= 0) & (gx < WDIM);
        if (inb) ldgsts_f32(&skel[off + (size_t)gy * WDIM + gx], &sS[i]);
        else     sS[i] = 0.0f;
    }
    wait_async_lds();
    __syncthreads();

    for (int i = tid; i < 16 * 512; i += 256) {
        const int ry = i >> 9, x = i & 511;
        const float c = sS[(ry + 1) * 514 + (x + 1)];
        float ns = 9.0f * c;
        for (int dy = 0; dy < 3; ++dy)
            for (int dx = 0; dx < 3; ++dx)
                ns += sS[(ry + dy) * 514 + (x + dx)];
        const float d = ns - 11.0f;
        sEp[i] = (f16_t)(expf(-d * d) * c);
    }
    __syncthreads();

    // --- WMMA weighted reduction: wave w covers columns [w*64, w*64+64) -----
    const int lane = tid & 31;
    const int wave = tid >> 5;
    const int m    = lane & 15;               // A row index / B column (N) index
    const int g    = lane >> 4;

    // Fixed B (both chunks): col0 = ones, col1 = local k within chunk.
    // B (32x16 f16): lanes 0-15 hold K=0..7,16..23 (N=lane), lanes 16-31 K=8..15,24..31.
    const float sel0 = (m == 0) ? 1.0f : 0.0f;
    const float sel1 = (m == 1) ? 1.0f : 0.0f;
    const int kb = g * 8;
    v16h bm;
#pragma unroll
    for (int i = 0; i < 16; ++i)
        bm[i] = (f16_t)(sel0 + sel1 * (float)(kb + i + (i >= 8 ? 8 : 0)));

    const int base0 = wave * 64;
    v8f acc[2] = { v8f{}, v8f{} };
#pragma unroll
    for (int cc = 0; cc < 2; ++cc) {
        const int base = base0 + cc * 32;        // K chunk of 32 columns
        // A (16x32 f16): ISA 7.12.2 layout — lane halves split K in 8s/16s
        v16h am;
#pragma unroll
        for (int v = 0; v < 8; ++v)
#pragma unroll
            for (int j = 0; j < 2; ++j) {
                const int K = 16 * (v >> 2) + 8 * g + 2 * (v & 3) + j;
                am[2 * v + j] = sEp[m * 512 + base + K];
            }
        acc[cc] = __builtin_amdgcn_wmma_f32_16x16x32_f16(
                      /*neg_a=*/false, am, /*neg_b=*/false, bm,
                      /*c_mod=*/(short)0, acc[cc],
                      /*reuse_a=*/false, /*reuse_b=*/false);
    }

    // D layout: VGPR r, lanes0-15 -> (M=r, N=lane), lanes16-31 -> (M=r+8, N=lane-16)
    if (m < 2) {
#pragma unroll
        for (int cc = 0; cc < 2; ++cc)
#pragma unroll
            for (int r = 0; r < 8; ++r)
                sRed[wave][cc][m][r + 8 * g] = acc[cc][r];
    }
    __syncthreads();

    if (tid == 0) {
        float S = 0.f, Sy = 0.f, Sx = 0.f;
        for (int row = 0; row < 16; ++row) {
            float rs = 0.f, rx = 0.f;
            for (int w = 0; w < 8; ++w) {
                const float a0 = sRed[w][0][0][row], a1 = sRed[w][1][0][row];
                const float t0 = sRed[w][0][1][row], t1 = sRed[w][1][1][row];
                rs += a0 + a1;
                rx += t0 + t1 + (float)(w * 64) * a0 + (float)(w * 64 + 32) * a1;
            }
            S += rs; Sx += rx; Sy += (float)(y0 + row) * rs;
        }
        epPart[blockIdx.x * 3 + 0] = S;
        epPart[blockIdx.x * 3 + 1] = Sy;
        epPart[blockIdx.x * 3 + 2] = Sx;
    }
}

// ---------------------------------------------------------------------------
// Kernel 5: final scalar assembly (deterministic, single block).
// ---------------------------------------------------------------------------
__global__ void final_kernel(const float* __restrict__ dicePart,
                             const float* __restrict__ epPart,
                             float* __restrict__ out) {
    const int tid = threadIdx.x;
    __shared__ float sm[256];
    __shared__ float dcS[3];
    __shared__ float imgS[NIMG][3];

    for (int c = 0; c < 3; ++c) {
        float a = 0.f;
        for (int i = tid; i < 1024; i += 256) a += dicePart[i * 3 + c];
        sm[tid] = a;
        __syncthreads();
        for (int s = 128; s > 0; s >>= 1) {
            if (tid < s) sm[tid] += sm[tid + s];
            __syncthreads();
        }
        if (tid == 0) dcS[c] = sm[0];
        __syncthreads();
    }

    if (tid < NIMG) {
        float S = 0.f, Sy = 0.f, Sx = 0.f;
        for (int s = 0; s < 32; ++s) {
            const float* p = &epPart[(tid * 32 + s) * 3];
            S += p[0]; Sy += p[1]; Sx += p[2];
        }
        imgS[tid][0] = S; imgS[tid][1] = Sy; imgS[tid][2] = Sx;
    }
    __syncthreads();

    if (tid == 0) {
        float sumDist = 0.f, cp = 0.f;
        for (int b = 0; b < BATCH; ++b) {
            const float Sp = imgS[b][0], Syp = imgS[b][1], Sxp = imgS[b][2];
            const float St = imgS[BATCH + b][0], Syt = imgS[BATCH + b][1], Sxt = imgS[BATCH + b][2];
            const float ycp = Syp / (Sp + 1e-8f), xcp = Sxp / (Sp + 1e-8f);
            const float yct = Syt / (St + 1e-8f), xct = Sxt / (St + 1e-8f);
            const float dy = ycp - yct, dx = xcp - xct;
            sumDist += sqrtf(dy * dy + dx * dx);
            cp += fabsf(Sp - St) / (Sp + St + 1e-8f);
        }
        const float diag = sqrtf((float)(HDIM * HDIM + WDIM * WDIM));
        const float distance_loss = (sumDist / (float)BATCH) / (diag * 1.0f + 1e-8f);
        const float count_penalty = cp / (float)BATCH;
        const float fel  = distance_loss + 1.0f * count_penalty;
        const float dice = 1.0f - (2.0f * dcS[0] + 1.0f) / (dcS[1] + dcS[2] + 1.0f);
        out[0] = 0.85f * dice + 0.15f * fel;
    }
}

// ---------------------------------------------------------------------------
extern "C" void kernel_launch(void* const* d_in, const int* in_sizes, int n_in,
                              void* d_out, int out_size, void* d_ws, size_t ws_size,
                              hipStream_t stream) {
    (void)in_sizes; (void)n_in; (void)out_size; (void)ws_size;
    const float* no = (const float*)d_in[0];
    const int*   yt = (const int*)d_in[1];

    float* ws      = (float*)d_ws;
    float* imgA    = ws;                                   // 16 * 262144
    float* imgB    = ws + (size_t)NIMG * HWSZ;             // 16 * 262144
    float* skel    = ws + 2 * (size_t)NIMG * HWSZ;         // 16 * 262144
    float* epPart  = ws + 3 * (size_t)NIMG * HWSZ;         // 512 * 3
    float* dicePart = epPart + 512 * 3;                    // 1024 * 3

    prep_kernel<<<1024, 256, 0, stream>>>(no, yt, imgA, dicePart);
    init_skel_kernel<<<NIMG * NTILES, 256, 0, stream>>>(imgA, skel);

    float* src = imgA;
    float* dst = imgB;
    for (int it = 0; it < NITER; ++it) {
        iter_kernel<<<NIMG * NTILES, 256, 0, stream>>>(src, dst, skel);
        float* t = src; src = dst; dst = t;
    }

    endpoint_kernel<<<NIMG * 32, 256, 0, stream>>>(skel, epPart);
    final_kernel<<<1, 256, 0, stream>>>(dicePart, epPart, (float*)d_out);
}
